// VisionAttention_17901423690400
// MI455X (gfx1250) — matrix operand order repaired and checked by reading, not compiled, not run on hardware
//
#include <hip/hip_runtime.h>
#include <hip/hip_bf16.h>

typedef __attribute__((ext_vector_type(16))) _Float16 v16h;
typedef __attribute__((ext_vector_type(8)))  float    v8f;

union F16x16 { v16h v; uint4 u[2]; _Float16 h[16]; };
union F16x8  { uint4 u; _Float16 h[8]; };

__device__ __forceinline__ v8f zv8() {
    v8f z = {0.f, 0.f, 0.f, 0.f, 0.f, 0.f, 0.f, 0.f};
    return z;
}

__device__ __forceinline__ v8f wmma_f16(v16h a, v16h b, v8f c) {
    // D(f32 16x16) = A(f16 16x32) * B(f16 32x16) + C
    return __builtin_amdgcn_wmma_f32_16x16x32_f16(
        /*neg_a=*/false, a, /*neg_b=*/false, b,
        /*c_mod=*/(short)0, c, /*reuse_a=*/false, /*reuse_b=*/false);
}

// ---------------------------------------------------------------------------
// f32 -> f16 conversion (grid-stride, vectorized x4)
// ---------------------------------------------------------------------------
__global__ void cvt_f32_f16(const float* __restrict__ in, _Float16* __restrict__ out, size_t n) {
    size_t i = (size_t)blockIdx.x * blockDim.x + threadIdx.x;
    size_t stride = (size_t)gridDim.x * blockDim.x;
    for (size_t j = i * 4; j < n; j += stride * 4) {
        float4 v = *(const float4*)(in + j);
        out[j + 0] = (_Float16)v.x;
        out[j + 1] = (_Float16)v.y;
        out[j + 2] = (_Float16)v.z;
        out[j + 3] = (_Float16)v.w;
    }
}

// ---------------------------------------------------------------------------
// WMMA GEMM: C[M,N] = A[M,K](f16) * B[K,N](f16) + bias, OutT in {f16, f32}.
// Block tile 128x128, 8 waves in a 4x2 grid; wave (wr,wc) owns a 32x64
// sub-tile = 2x4 WMMA accumulators (B frags reused 2x, A frags 4x).
// Software-pipelined: K-step n+1 global loads issued while K-step n computes;
// K-step n+2 prefetched (global_prefetch_b8).
// LDS: A row-major [128][32]; B transposed [n][k] so B fragments are
// two 8-K chunks per lane (k = 8h.., 16+8h..), matching the A fragments.
// ---------------------------------------------------------------------------
template <typename OutT>
__global__ __launch_bounds__(256) void gemm128x128(
    const _Float16* __restrict__ A, const _Float16* __restrict__ B,
    const float* __restrict__ bias, OutT* __restrict__ C, int K, int N)
{
    __shared__ __align__(16) _Float16 sA[128][32];
    __shared__ __align__(16) _Float16 sBt[128][32];

    const int tid  = threadIdx.x;
    const int lane = tid & 31;
    const int w    = tid >> 5;
    const int wr   = w >> 1;          // 0..3 : row group (32 rows each)
    const int wc   = w & 1;           // 0..1 : col group (64 cols each)
    const int m0   = blockIdx.y * 128;
    const int n0   = blockIdx.x * 128;
    const int mm   = lane & 15;       // tile row / col within 16
    const int hi   = lane >> 4;       // half-wave selector

    v8f acc[2][4];
    #pragma unroll
    for (int mt = 0; mt < 2; ++mt)
        #pragma unroll
        for (int t = 0; t < 4; ++t) acc[mt][t] = zv8();

    // global staging assignments (256 threads, 16 f16 each for A and B)
    const int arow  = tid >> 1;              // 0..127
    const int ahalf = (tid & 1) * 16;        // 0 or 16
    const _Float16* aptr = A + (size_t)(m0 + arow) * K + ahalf;
    const int bk = tid >> 3;                 // 0..31
    const int bn = (tid & 7) * 16;           // 0..112
    const _Float16* bptr = B + (size_t)bk * N + n0 + bn;

    // preload K-step 0
    uint4 a0 = *(const uint4*)(aptr);
    uint4 a1 = *(const uint4*)(aptr + 8);
    F16x8 b0, b1;
    b0.u = *(const uint4*)(bptr);
    b1.u = *(const uint4*)(bptr + 8);

    for (int kb = 0; kb < K; kb += 32) {
        // commit staged registers to LDS
        *(uint4*)&sA[arow][ahalf]     = a0;
        *(uint4*)&sA[arow][ahalf + 8] = a1;
        #pragma unroll
        for (int i = 0; i < 8; ++i) {
            sBt[bn + i][bk]     = b0.h[i];
            sBt[bn + 8 + i][bk] = b1.h[i];
        }
        __syncthreads();

        // issue next K-step global loads (overlap with WMMA below)
        if (kb + 32 < K) {
            a0 = *(const uint4*)(aptr + kb + 32);
            a1 = *(const uint4*)(aptr + kb + 40);
            b0.u = *(const uint4*)(bptr + (size_t)(kb + 32) * N);
            b1.u = *(const uint4*)(bptr + (size_t)(kb + 32) * N + 8);
        }
        if (kb + 64 < K) {
            __builtin_prefetch(aptr + kb + 64, 0, 0);
            __builtin_prefetch(bptr + (size_t)(kb + 64) * N, 0, 0);
        }

        // A fragments for this wave's two 16-row sub-tiles
        F16x16 af[2];
        #pragma unroll
        for (int mt = 0; mt < 2; ++mt) {
            af[mt].u[0] = *(const uint4*)&sA[wr * 32 + mt * 16 + mm][hi * 8];
            af[mt].u[1] = *(const uint4*)&sA[wr * 32 + mt * 16 + mm][16 + hi * 8];
        }
        // 4 B fragments, each reused across 2 WMMAs
        #pragma unroll
        for (int t = 0; t < 4; ++t) {
            F16x16 bf;
            const _Float16* p = &sBt[wc * 64 + t * 16 + mm][hi * 8];
            bf.u[0] = *(const uint4*)p;
            bf.u[1] = *(const uint4*)(p + 16);
            #pragma unroll
            for (int mt = 0; mt < 2; ++mt)
                acc[mt][t] = wmma_f16(af[mt].v, bf.v, acc[mt][t]);
        }
        __syncthreads();
    }

    // epilogue: C/D layout col = lane&15, row = r + 8*(lane>>4)
    #pragma unroll
    for (int mt = 0; mt < 2; ++mt) {
        OutT* crow = C + (size_t)(m0 + wr * 32 + mt * 16) * N + n0 + wc * 64;
        #pragma unroll
        for (int t = 0; t < 4; ++t) {
            float bv = bias[n0 + wc * 64 + t * 16 + mm];
            #pragma unroll
            for (int r = 0; r < 8; ++r) {
                int m = r + 8 * hi;
                crow[(size_t)m * N + t * 16 + mm] = (OutT)(acc[mt][t][r] + bv);
            }
        }
    }
}

// ---------------------------------------------------------------------------
// RoPE in place on q and k halves of qkv (f16). Each thread owns the pair
// (d, d+40) of one (s, h) so the in-place update is race-free.
// ---------------------------------------------------------------------------
__global__ void rope_inplace(_Float16* __restrict__ qkv,
                             const float* __restrict__ cosb,
                             const float* __restrict__ sinb, int S)
{
    int idx = blockIdx.x * blockDim.x + threadIdx.x;
    int total = S * 16 * 40;
    if (idx >= total) return;
    int d = idx % 40;
    int t = idx / 40;
    int h = t % 16;
    int s = t / 16;
    float c0 = cosb[s * 80 + d], c1 = cosb[s * 80 + d + 40];
    float s0 = sinb[s * 80 + d], s1 = sinb[s * 80 + d + 40];
    size_t base = (size_t)s * 3840 + h * 80 + d;
    // q
    float q0 = (float)qkv[base], q1 = (float)qkv[base + 40];
    qkv[base]      = (_Float16)(q0 * c0 - q1 * s0);
    qkv[base + 40] = (_Float16)(q1 * c1 + q0 * s1);
    // k (offset 1280)
    float k0 = (float)qkv[base + 1280], k1 = (float)qkv[base + 1320];
    qkv[base + 1280] = (_Float16)(k0 * c0 - k1 * s0);
    qkv[base + 1320] = (_Float16)(k1 * c1 + k0 * s1);
}

// ---------------------------------------------------------------------------
// Windowed attention: one block per (window, head). 4 waves; wave w owns
// query rows [16w,16w+16). scores 64x64 = 4x4 WMMA tiles with K=80 padded
// to 96; softmax via __shfl_xor over the 16-lane column groups (wave32);
// out [64x80] = P[64x64] @ V[64x80] as 5 N-tiles, K=64 (2 steps).
// ---------------------------------------------------------------------------
__global__ __launch_bounds__(128) void attn_win64(
    const _Float16* __restrict__ qkv, _Float16* __restrict__ out, int S)
{
    __shared__ __align__(16) _Float16 sK[64][96];    // key-major, d padded to 96
    __shared__ __align__(16) _Float16 sVt[80][64];   // d-major (B layout for PV)
    __shared__ __align__(16) _Float16 sP[4][16][64]; // per-wave probabilities

    const int tid  = threadIdx.x;
    const int lane = tid & 31;
    const int w    = tid >> 5;
    const int win  = blockIdx.x >> 4;
    const int h    = blockIdx.x & 15;
    const int mm   = lane & 15;
    const int hi   = lane >> 4;
    const size_t ld = 3840;
    (void)S;

    const _Float16* kg = qkv + (size_t)win * 64 * ld + 1280 + h * 80;
    const _Float16* vg = qkv + (size_t)win * 64 * ld + 2560 + h * 80;

    // stage K tile [64][80] (16B chunks) + zero-pad d=80..95
    for (int c = tid; c < 640; c += 128) {
        int row = c / 10, part = (c % 10) * 8;
        *(uint4*)&sK[row][part] = *(const uint4*)(kg + (size_t)row * ld + part);
    }
    {
        uint4 z = {0, 0, 0, 0};
        int row = tid >> 1, part = 80 + (tid & 1) * 8;  // 128 threads cover 64x2 chunks
        *(uint4*)&sK[row][part] = z;
    }
    // stage V transposed: sVt[d][key]
    for (int c = tid; c < 640; c += 128) {
        int key = c / 10, part = (c % 10) * 8;
        F16x8 t8;
        t8.u = *(const uint4*)(vg + (size_t)key * ld + part);
        #pragma unroll
        for (int i = 0; i < 8; ++i) sVt[part + i][key] = t8.h[i];
    }
    __syncthreads();

    // ---- scores = Q @ K^T -------------------------------------------------
    v8f sc[4];
    #pragma unroll
    for (int t = 0; t < 4; ++t) sc[t] = zv8();

    const _Float16* qrow = qkv + (size_t)(win * 64 + w * 16 + mm) * ld + h * 80;
    #pragma unroll
    for (int ks = 0; ks < 3; ++ks) {
        F16x16 af;
        int d0 = ks * 32 + hi * 8;
        int d1 = ks * 32 + 16 + hi * 8;
        af.u[0] = *(const uint4*)(qrow + d0);
        if (d1 < 80) {
            af.u[1] = *(const uint4*)(qrow + d1);
        } else {
            uint4 z = {0, 0, 0, 0};
            af.u[1] = z;
        }
        #pragma unroll
        for (int t = 0; t < 4; ++t) {
            F16x16 bf;
            const _Float16* p = &sK[t * 16 + mm][ks * 32 + hi * 8];
            bf.u[0] = *(const uint4*)p;
            bf.u[1] = *(const uint4*)(p + 16);
            sc[t] = wmma_f16(af.v, bf.v, sc[t]);
        }
    }

    // ---- softmax over 64 keys per query row -------------------------------
    // lane holds rows m = r + 8*hi, key = t*16 + mm; reduce over mm (16 lanes)
    const float scale = 0.11180339887498949f;  // 80^-0.5
    float ex[4][8];
    float rinv[8];
    #pragma unroll
    for (int r = 0; r < 8; ++r) {
        float m = -1e30f;
        #pragma unroll
        for (int t = 0; t < 4; ++t) m = fmaxf(m, sc[t][r]);
        #pragma unroll
        for (int off = 8; off >= 1; off >>= 1) m = fmaxf(m, __shfl_xor(m, off, 32));
        float rmax = m * scale;
        float sum = 0.f;
        #pragma unroll
        for (int t = 0; t < 4; ++t) {
            float e = __expf(sc[t][r] * scale - rmax);
            ex[t][r] = e;
            sum += e;
        }
        #pragma unroll
        for (int off = 8; off >= 1; off >>= 1) sum += __shfl_xor(sum, off, 32);
        rinv[r] = 1.f / sum;
    }
    #pragma unroll
    for (int t = 0; t < 4; ++t) {
        #pragma unroll
        for (int r = 0; r < 8; ++r)
            sP[w][r + 8 * hi][t * 16 + mm] = (_Float16)(ex[t][r] * rinv[r]);
    }
    __syncthreads();

    // ---- out = P @ V ------------------------------------------------------
    v8f oa[5];
    #pragma unroll
    for (int t = 0; t < 5; ++t) oa[t] = zv8();
    #pragma unroll
    for (int ks = 0; ks < 2; ++ks) {
        F16x16 af;
        af.u[0] = *(const uint4*)&sP[w][mm][ks * 32 + hi * 8];
        af.u[1] = *(const uint4*)&sP[w][mm][ks * 32 + 16 + hi * 8];
        #pragma unroll
        for (int nt = 0; nt < 5; ++nt) {
            F16x16 bf;
            const _Float16* p = &sVt[nt * 16 + mm][ks * 32 + hi * 8];
            bf.u[0] = *(const uint4*)p;
            bf.u[1] = *(const uint4*)(p + 16);
            oa[nt] = wmma_f16(af.v, bf.v, oa[nt]);
        }
    }
    _Float16* ob = out + (size_t)(win * 64 + w * 16) * 1280 + h * 80;
    #pragma unroll
    for (int nt = 0; nt < 5; ++nt) {
        #pragma unroll
        for (int r = 0; r < 8; ++r)
            ob[(size_t)(r + 8 * hi) * 1280 + nt * 16 + mm] = (_Float16)oa[nt][r];
    }
}

// ---------------------------------------------------------------------------
extern "C" void kernel_launch(void* const* d_in, const int* in_sizes, int n_in,
                              void* d_out, int out_size, void* d_ws, size_t ws_size,
                              hipStream_t stream) {
    const float* x     = (const float*)d_in[0];
    const float* cosb  = (const float*)d_in[1];
    const float* sinb  = (const float*)d_in[2];
    // d_in[3] = cu_seqlens: uniform 64-token windows (per reference), unused
    const float* Wqkv  = (const float*)d_in[4];
    const float* bqkv  = (const float*)d_in[5];
    const float* Wproj = (const float*)d_in[6];
    const float* bproj = (const float*)d_in[7];
    float* out = (float*)d_out;
    (void)n_in; (void)out_size; (void)ws_size;

    const int HID = 1280, N3 = 3840;
    const int S = in_sizes[0] / HID;  // 32768; multiple of 128

    char* wp = (char*)d_ws;
    auto take = [&](size_t bytes) {
        char* p = wp;
        wp += (bytes + 255) & ~(size_t)255;
        return p;
    };
    _Float16* xh   = (_Float16*)take((size_t)S * HID * 2);
    _Float16* Wqh  = (_Float16*)take((size_t)HID * N3 * 2);
    _Float16* Wph  = (_Float16*)take((size_t)HID * HID * 2);
    _Float16* qkvh = (_Float16*)take((size_t)S * N3 * 2);
    _Float16* atth = (_Float16*)take((size_t)S * HID * 2);

    cvt_f32_f16<<<2048, 256, 0, stream>>>(x, xh, (size_t)S * HID);
    cvt_f32_f16<<<2048, 256, 0, stream>>>(Wqkv, Wqh, (size_t)HID * N3);
    cvt_f32_f16<<<2048, 256, 0, stream>>>(Wproj, Wph, (size_t)HID * HID);

    // qkv = x @ Wqkv + bias   (f16 out, f32 accumulate)
    gemm128x128<_Float16><<<dim3(N3 / 128, S / 128), 256, 0, stream>>>(
        xh, Wqh, bqkv, qkvh, HID, N3);

    // RoPE on q,k in place
    int nrope = S * 16 * 40;
    rope_inplace<<<(nrope + 255) / 256, 256, 0, stream>>>(qkvh, cosb, sinb, S);

    // windowed attention: one block per (window, head)
    attn_win64<<<(S / 64) * 16, 128, 0, stream>>>(qkvh, atth, S);

    // out = attn @ Wproj + bias  (f32 out)
    gemm128x128<float><<<dim3(HID / 128, S / 128), 256, 0, stream>>>(
        atth, Wph, bproj, out, HID, HID);
}
